// SelfAttention_41841571397642
// MI455X (gfx1250) — matrix operand order repaired and checked by reading, not compiled, not run on hardware
//
#include <hip/hip_runtime.h>
#include <hip/hip_bf16.h>
#include <cstdint>

// ---------------------------------------------------------------------------
// Self-attention block (GN -> QKV -> softmax(QK^T/sqrt(C))V -> proj -> +x)
// B=4, C=256, N=H*W=4096.  ~77 GFLOP vs ~32MB mandatory traffic => compute
// bound => everything on v_wmma_f32_16x16x32_f16.  Flash-attention structure;
// K tiles staged in LDS via the Tensor Data Mover (double buffered, padded
// rows for bank-friendly ds_load_b128), synced with s_wait_tensorcnt.
// ---------------------------------------------------------------------------

typedef __attribute__((ext_vector_type(16))) _Float16 v16h;
typedef __attribute__((ext_vector_type(8)))  _Float16 v8h;
typedef __attribute__((ext_vector_type(8)))  float    v8f;
typedef __attribute__((ext_vector_type(4)))  unsigned int v4u;
typedef __attribute__((ext_vector_type(8)))  int      v8i;
typedef __attribute__((ext_vector_type(4)))  int      v4i;

namespace cfg {
constexpr int Bc = 4;
constexpr int C  = 256;
constexpr int N  = 4096;     // H*W
constexpr int G  = 32;       // groups
constexpr int CG = C / G;    // 8 channels per group
constexpr int KLD = 272;     // K-tile LDS row stride in halves (256 + 16 pad)
}

#if __has_builtin(__builtin_amdgcn_tensor_load_to_lds)
#define USE_TDM 1
#else
#define USE_TDM 0
#endif

// --- WMMA helpers -----------------------------------------------------------

__device__ __forceinline__ v8f wmma_f16(v16h a, v16h b, v8f c) {
  return __builtin_amdgcn_wmma_f32_16x16x32_f16(false, a, false, b, (short)0, c,
                                                false, false);
}

// A-matrix fragment (16x32, f16). Row-major source: row = row0 + (lane&15).
// Per-lane K pattern: elements 0..7 = k0 + h*8 + i, 8..15 = k0 + 16 + h*8 + i.
__device__ __forceinline__ v16h load_a_frag(const _Float16* base, int ld,
                                            int row0, int k0, int lane) {
  int r = lane & 15, h = (lane >> 4) & 1;
  const _Float16* p = base + (size_t)(row0 + r) * ld + k0 + h * 8;
  v8h lo = *(const v8h*)p;
  v8h hi = *(const v8h*)(p + 16);
  v16h out;
#pragma unroll
  for (int i = 0; i < 8; i++) { out[i] = lo[i]; out[i + 8] = hi[i]; }
  return out;
}

// B-matrix fragment (32x16, f16) from B^T rows (row = column of B).
// Per-lane: column = col0 + (lane&15), elements 0..7 = k0 + h*8 + i, 8..15 = k0 + 16 + h*8 + i.
__device__ __forceinline__ v16h load_b_frag(const _Float16* bt, int ld,
                                            int col0, int k0, int lane) {
  int n = lane & 15, h = (lane >> 4) & 1;
  const _Float16* p = bt + (size_t)(col0 + n) * ld + k0 + h * 8;
  v8h lo = *(const v8h*)p;
  v8h hi = *(const v8h*)(p + 16);
  v16h out;
#pragma unroll
  for (int i = 0; i < 8; i++) { out[i] = lo[i]; out[i + 8] = hi[i]; }
  return out;
}

#if USE_TDM
// TDM: async-copy one 32x256 f16 K-tile (16KB contiguous in global) into LDS,
// inserting 32B of padding per 512B row (-> [32][272] half layout).
// D# packing per CDNA5 ISA.
__device__ __forceinline__ void tdm_load_ktile(const _Float16* gsrc,
                                               unsigned lds_byte_addr) {
  unsigned long long ga = (unsigned long long)(uintptr_t)gsrc;
  v4u g0;
  g0[0] = 1u;                                   // count=1 (valid), user mode
  g0[1] = lds_byte_addr;                        // lds_addr
  g0[2] = (unsigned)ga;                         // global_addr[31:0]
  g0[3] = (unsigned)((ga >> 32) & 0x01FFFFFFu)  // global_addr[56:32]
          | 0x80000000u;                        // type=2 at bits[127:126]
  v8i g1;
  g1[0] = (int)((1u << 16)      // data_size = 1 (2 bytes)
                | (1u << 20)    // pad_enable
                | (6u << 22)    // pad_interval: every 128 DWORDs (512B)
                | (7u << 25));  // pad_amount: 8 DWORDs (32B)
  g1[1] = (int)(0x2000u << 16); // tensor_dim0[15:0] = 8192 @ bits[63:48]
  g1[2] = (int)(1u << 16);      // tensor_dim1[15:0] = 1    @ bits[95:80]
  g1[3] = (int)(0x2000u << 16); // tile_dim0 = 8192         @ bits[127:112]
  g1[4] = 1;                    // tile_dim1 = 1
  g1[5] = 0x2000;               // tensor_dim0_stride[31:0] = 8192
  g1[6] = 0;
  g1[7] = 0;
  v4i z = {0, 0, 0, 0};
#if defined(__clang_major__) && (__clang_major__ >= 23)
  v8i z8 = {0, 0, 0, 0, 0, 0, 0, 0};
  __builtin_amdgcn_tensor_load_to_lds(g0, g1, z, z, z8, 0);
#else
  __builtin_amdgcn_tensor_load_to_lds(g0, g1, z, z, 0);
#endif
}
#endif

// --- K0: f32 -> f16 weight conversion ---------------------------------------

__global__ void cvt_f32_f16_kernel(const float* __restrict__ src,
                                   _Float16* __restrict__ dst, int n) {
  int i = blockIdx.x * 256 + threadIdx.x;
  if (i < n) dst[i] = (_Float16)src[i];
}

// --- K1: GroupNorm statistics (one block per (b, group)) --------------------

__global__ __launch_bounds__(256) void gn_stats_kernel(
    const float* __restrict__ x, float* __restrict__ stats) {
  using namespace cfg;
  int bg = blockIdx.x;                       // b*G + g
  const float* p = x + (size_t)bg * CG * N;  // group channels are contiguous
  float s = 0.f, s2 = 0.f;
  for (int i = threadIdx.x; i < CG * N; i += 256) {
    float v = p[i];
    s += v; s2 += v * v;
  }
  int lane = threadIdx.x & 31, w = threadIdx.x >> 5;
#pragma unroll
  for (int m = 16; m >= 1; m >>= 1) {
    s  += __shfl_xor(s,  m, 32);
    s2 += __shfl_xor(s2, m, 32);
  }
  __shared__ float red[2][8];
  if (lane == 0) { red[0][w] = s; red[1][w] = s2; }
  __syncthreads();
  if (threadIdx.x == 0) {
    float S = 0.f, S2 = 0.f;
#pragma unroll
    for (int i = 0; i < 8; i++) { S += red[0][i]; S2 += red[1][i]; }
    const float inv = 1.0f / (float)(CG * N);
    float mean = S * inv;
    float var  = S2 * inv - mean * mean;
    stats[bg]          = mean;
    stats[Bc * G + bg] = rsqrtf(var + 1e-5f);
  }
}

// --- K2: GN apply + transpose to h[B, N, C] (f16) ---------------------------

__global__ __launch_bounds__(256) void gn_apply_kernel(
    const float* __restrict__ x, const float* __restrict__ gamma,
    const float* __restrict__ beta, const float* __restrict__ stats,
    _Float16* __restrict__ hbuf) {
  using namespace cfg;
  __shared__ float tile[64][65];
  int n0 = blockIdx.x * 64, c0 = blockIdx.y * 64, b = blockIdx.z;
  int t = threadIdx.x;
  const float* xb = x + (size_t)b * C * N;
#pragma unroll
  for (int i = 0; i < 16; i++) {
    int idx = t + 256 * i;
    int cl = idx >> 6, nl = idx & 63;     // consecutive t -> consecutive n
    int c = c0 + cl;
    int g = c >> 3;
    float mean = stats[b * G + g];
    float rstd = stats[Bc * G + b * G + g];
    float v = xb[(size_t)c * N + n0 + nl];
    tile[cl][nl] = (v - mean) * rstd * gamma[c] + beta[c];
  }
  __syncthreads();
  _Float16* hb = hbuf + (size_t)b * N * C;
#pragma unroll
  for (int i = 0; i < 16; i++) {
    int idx = t + 256 * i;
    int nl = idx >> 6, cl = idx & 63;     // consecutive t -> consecutive c
    hb[(size_t)(n0 + nl) * C + c0 + cl] = (_Float16)tile[cl][nl];
  }
}

// --- K3: Q/K/V projections (WMMA GEMM, M=positions) -------------------------
// q/k stored [B,N,C] f16 (A-rows for attention); v stored transposed [B,C,N].

__global__ __launch_bounds__(256) void qkv_proj_kernel(
    const _Float16* __restrict__ hbuf, const _Float16* __restrict__ wqh,
    const _Float16* __restrict__ wkh, const _Float16* __restrict__ wvh,
    const float* __restrict__ bq, const float* __restrict__ bk,
    const float* __restrict__ bv, _Float16* __restrict__ qbuf,
    _Float16* __restrict__ kbuf, _Float16* __restrict__ vt) {
  using namespace cfg;
  int z = blockIdx.z;
  int b = z / 3, which = z % 3;
  int m0 = blockIdx.x * 64;   // position tile
  int n0 = blockIdx.y * 32;   // out-channel tile
  int w = threadIdx.x >> 5, lane = threadIdx.x & 31;
  int mrow = m0 + (w & 3) * 16;
  int ncol = n0 + (w >> 2) * 16;

  const _Float16* A  = hbuf + (size_t)b * N * C;
  const _Float16* Bt = (which == 0) ? wqh : (which == 1) ? wkh : wvh;
  const float* bias  = (which == 0) ? bq  : (which == 1) ? bk  : bv;

  v8f acc = {};
#pragma unroll
  for (int k = 0; k < C; k += 32) {
    v16h a  = load_a_frag(A, C, mrow, k, lane);
    v16h bb = load_b_frag(Bt, C, ncol, k, lane);
    acc = wmma_f16(a, bb, acc);
  }
  int r = lane & 15, h = (lane >> 4) & 1;
  int c = ncol + r;
  float bi = bias[c];
#pragma unroll
  for (int v = 0; v < 8; v++) {
    int m = mrow + v + 8 * h;
    float val = acc[v] + bi;
    if (which == 0) {
      qbuf[((size_t)b * N + m) * C + c] = (_Float16)(val * 0.0625f);  // 1/sqrt(C)
    } else if (which == 1) {
      kbuf[((size_t)b * N + m) * C + c] = (_Float16)val;
    } else {
      vt[((size_t)b * C + c) * N + m] = (_Float16)val;                // transposed
    }
  }
}

// --- K4: flash attention ----------------------------------------------------
// Wave owns 16 query rows; K streamed in 32-key tiles staged in LDS by the
// TDM (double buffered, async, s_wait_tensorcnt) while WMMAs run; online
// softmax with 16-lane xor reductions; P transposed D->A layout through LDS.

__global__ __launch_bounds__(256) void attn_kernel(
    const _Float16* __restrict__ qbuf, const _Float16* __restrict__ kbuf,
    const _Float16* __restrict__ vt, _Float16* __restrict__ obuf) {
  using namespace cfg;
  __shared__ _Float16 ktile[2][32][KLD];   // 2 x (32 keys x 256 ch), padded
  __shared__ _Float16 pbuf[8][16][40];     // 16x32 P per wave, padded rows
  int b = blockIdx.y;
  int w = threadIdx.x >> 5, lane = threadIdx.x & 31;
  int qrow = blockIdx.x * 128 + w * 16;
  int r = lane & 15, h = (lane >> 4) & 1;

  const _Float16* Q = qbuf + (size_t)b * N * C;
  const _Float16* K = kbuf + (size_t)b * N * C;
  const _Float16* V = vt   + (size_t)b * C * N;

  v16h qf[8];
#pragma unroll
  for (int k = 0; k < 8; k++) qf[k] = load_a_frag(Q, C, qrow, k * 32, lane);

  v8f o[16];
#pragma unroll
  for (int i = 0; i < 16; i++) o[i] = (v8f){};
  float mrun[8], lrun[8];
#pragma unroll
  for (int v = 0; v < 8; v++) { mrun[v] = -3.0e38f; lrun[v] = 0.f; }

  // prologue: stage K tile 0 into buffer 0
#if USE_TDM
  if (w == 0) {
    tdm_load_ktile(K, (unsigned)(uintptr_t)&ktile[0][0][0]);
    __builtin_amdgcn_s_wait_tensorcnt(0);
  }
#else
  for (int i = 0; i < 4; i++) {
    int e = (i * 256 + threadIdx.x) * 8;
    *(v8h*)&ktile[0][e >> 8][e & 255] = *(const v8h*)&K[e];
  }
#endif
  __syncthreads();

  for (int jt = 0; jt < N / 32; jt++) {
    int j = jt * 32, buf = jt & 1;

    // kick off async copy of the next K tile into the other buffer
#if USE_TDM
    if (w == 0 && jt + 1 < N / 32)
      tdm_load_ktile(K + (size_t)(j + 32) * C,
                     (unsigned)(uintptr_t)&ktile[buf ^ 1][0][0]);
#else
    if (jt + 1 < N / 32) {
      const _Float16* src = K + (size_t)(j + 32) * C;
      for (int i = 0; i < 4; i++) {
        int e = (i * 256 + threadIdx.x) * 8;
        *(v8h*)&ktile[buf ^ 1][e >> 8][e & 255] = *(const v8h*)&src[e];
      }
    }
#endif

    // S = Q K^T : two 16x16 tiles, K-dim 256, B-fragments from LDS
    const _Float16* kb = &ktile[buf][0][0];
    v8f s0 = {}, s1 = {};
#pragma unroll
    for (int k = 0; k < 8; k++) {
      v16h b0 = load_b_frag(kb, KLD, 0,  k * 32, lane);
      v16h b1 = load_b_frag(kb, KLD, 16, k * 32, lane);
      s0 = wmma_f16(qf[k], b0, s0);
      s1 = wmma_f16(qf[k], b1, s1);
    }

    // online softmax (row = v + 8*h, cols across 16-lane halves)
    float alpha[8];
#pragma unroll
    for (int v = 0; v < 8; v++) {
      float t = fmaxf(s0[v], s1[v]);
#pragma unroll
      for (int m = 8; m >= 1; m >>= 1) t = fmaxf(t, __shfl_xor(t, m, 32));
      float mn = fmaxf(mrun[v], t);
      alpha[v] = __expf(mrun[v] - mn);
      mrun[v] = mn;
      float p0 = __expf(s0[v] - mn);
      float p1 = __expf(s1[v] - mn);
      s0[v] = p0; s1[v] = p1;
      float ls = p0 + p1;
#pragma unroll
      for (int m = 8; m >= 1; m >>= 1) ls += __shfl_xor(ls, m, 32);
      lrun[v] = lrun[v] * alpha[v] + ls;
    }
#pragma unroll
    for (int cb = 0; cb < 16; cb++)
#pragma unroll
      for (int v = 0; v < 8; v++) o[cb][v] *= alpha[v];

    // P: D-layout -> LDS -> A-layout (intra-wave, dscnt wait only)
#pragma unroll
    for (int v = 0; v < 8; v++) {
      pbuf[w][v + 8 * h][r]      = (_Float16)s0[v];
      pbuf[w][v + 8 * h][16 + r] = (_Float16)s1[v];
    }
    asm volatile("s_wait_dscnt 0" ::: "memory");
    v16h pa;
    {
      const _Float16* pr = &pbuf[w][r][0];
#pragma unroll
      for (int i = 0; i < 8; i++) {
        pa[i]     = pr[h * 8 + i];
        pa[i + 8] = pr[16 + h * 8 + i];
      }
    }

    // O += P @ V  (V^T rows are channels -> contiguous B-fragments)
#pragma unroll
    for (int cb = 0; cb < 16; cb++) {
      v16h vb = load_b_frag(V, N, cb * 16, j, lane);
      o[cb] = wmma_f16(pa, vb, o[cb]);
    }

    // next tile ready + everyone done reading the current buffer
#if USE_TDM
    if (w == 0) __builtin_amdgcn_s_wait_tensorcnt(0);
#endif
    __syncthreads();
  }

  // epilogue: O /= l, store f16 [B,N,C]
#pragma unroll
  for (int v = 0; v < 8; v++) {
    float inv = 1.0f / lrun[v];
    int m = qrow + v + 8 * h;
#pragma unroll
    for (int cb = 0; cb < 16; cb++) {
      obuf[((size_t)b * N + m) * C + cb * 16 + r] = (_Float16)(o[cb][v] * inv);
    }
  }
}

// --- K5: output projection + bias + residual, store [B,C,N] -----------------
// M = out-channels (A = Wo rows), N = positions (Bt = obuf rows) so the
// [B,C,N] f32 store is coalesced across lanes.

__global__ __launch_bounds__(256) void out_proj_kernel(
    const _Float16* __restrict__ woh, const _Float16* __restrict__ obuf,
    const float* __restrict__ bo, const float* __restrict__ x,
    float* __restrict__ out) {
  using namespace cfg;
  int b = blockIdx.z;
  int pos0 = blockIdx.x * 32;
  int ch0  = blockIdx.y * 64;
  int w = threadIdx.x >> 5, lane = threadIdx.x & 31;
  int mrow = ch0  + (w & 3) * 16;   // channel
  int ncol = pos0 + (w >> 2) * 16;  // position
  const _Float16* Bt = obuf + (size_t)b * N * C;

  v8f acc = {};
#pragma unroll
  for (int k = 0; k < C; k += 32) {
    v16h a  = load_a_frag(woh, C, mrow, k, lane);
    v16h bb = load_b_frag(Bt, C, ncol, k, lane);
    acc = wmma_f16(a, bb, acc);
  }
  int r = lane & 15, h = (lane >> 4) & 1;
  int n = ncol + r;
#pragma unroll
  for (int v = 0; v < 8; v++) {
    int c = mrow + v + 8 * h;
    size_t idx = ((size_t)b * C + c) * N + n;
    out[idx] = acc[v] + bo[c] + x[idx];
  }
}

// ---------------------------------------------------------------------------

extern "C" void kernel_launch(void* const* d_in, const int* in_sizes, int n_in,
                              void* d_out, int out_size, void* d_ws, size_t ws_size,
                              hipStream_t stream) {
  using namespace cfg;
  const float* x     = (const float*)d_in[0];
  const float* gamma = (const float*)d_in[1];
  const float* beta  = (const float*)d_in[2];
  const float* wq = (const float*)d_in[3];  const float* bq = (const float*)d_in[4];
  const float* wk = (const float*)d_in[5];  const float* bk = (const float*)d_in[6];
  const float* wv = (const float*)d_in[7];  const float* bv = (const float*)d_in[8];
  const float* wo = (const float*)d_in[9];  const float* bo = (const float*)d_in[10];
  float* out = (float*)d_out;

  char* wsb = (char*)d_ws;
  float*    stats = (float*)wsb;                                  // 256 f32
  _Float16* wqh   = (_Float16*)(wsb + 1024);
  _Float16* wkh   = wqh + (size_t)C * C;
  _Float16* wvh   = wkh + (size_t)C * C;
  _Float16* woh   = wvh + (size_t)C * C;
  _Float16* hbuf  = (_Float16*)(wsb + (1u << 20));                // 8 MB
  _Float16* qbuf  = hbuf + (size_t)Bc * N * C;                    // 8 MB
  _Float16* kbuf  = qbuf + (size_t)Bc * N * C;                    // 8 MB
  _Float16* vt    = kbuf + (size_t)Bc * N * C;                    // 8 MB
  _Float16* obuf  = hbuf;  // hbuf dead after qkv_proj -> reuse

  const int wn = C * C;  // 65536
  cvt_f32_f16_kernel<<<wn / 256, 256, 0, stream>>>(wq, wqh, wn);
  cvt_f32_f16_kernel<<<wn / 256, 256, 0, stream>>>(wk, wkh, wn);
  cvt_f32_f16_kernel<<<wn / 256, 256, 0, stream>>>(wv, wvh, wn);
  cvt_f32_f16_kernel<<<wn / 256, 256, 0, stream>>>(wo, woh, wn);

  gn_stats_kernel<<<Bc * G, 256, 0, stream>>>(x, stats);
  gn_apply_kernel<<<dim3(N / 64, C / 64, Bc), 256, 0, stream>>>(
      x, gamma, beta, stats, hbuf);

  qkv_proj_kernel<<<dim3(N / 64, C / 32, Bc * 3), 256, 0, stream>>>(
      hbuf, wqh, wkh, wvh, bq, bk, bv, qbuf, kbuf, vt);

  attn_kernel<<<dim3(N / 128, Bc), 256, 0, stream>>>(qbuf, kbuf, vt, obuf);

  out_proj_kernel<<<dim3(N / 32, C / 64, Bc), 256, 0, stream>>>(
      woh, obuf, bo, x, out);
}
